// SelfAttention_65962107732181
// MI455X (gfx1250) — matrix operand order repaired and checked by reading, not compiled, not run on hardware
//
#include <hip/hip_runtime.h>
#include <math.h>

// ---------- types ----------
typedef _Float16 half_t;
typedef __attribute__((ext_vector_type(2)))  _Float16 half2_t;
typedef __attribute__((ext_vector_type(16))) _Float16 v16h;
typedef __attribute__((ext_vector_type(8)))  float    v8f;

#define WMMA_F16(a, b, c) \
  __builtin_amdgcn_wmma_f32_16x16x32_f16(false, (a), false, (b), (short)0, (c), false, false)

// K-offset pattern for the 16-bit A-matrix 16x32 fragment (CDNA5 ISA 7.12.2):
// VGPR v (<4): K = 2v + 8*half ; v in 4..7: K = 16 + 2(v-4) + 8*half (pairs of halves).
static __device__ __forceinline__ int a_frag_kbase(int v, int hf) {
  return (v < 4) ? (2 * v + 8 * hf) : (16 + 2 * (v - 4) + 8 * hf);
}

// ---------------- Tensor Data Mover (gfx1250) ----------------
#if __has_builtin(__builtin_amdgcn_tensor_load_to_lds)
#define HAVE_TDM 1
typedef __attribute__((ext_vector_type(4))) unsigned int u32x4;
typedef __attribute__((ext_vector_type(4))) int          i32x4;
typedef __attribute__((ext_vector_type(8))) int          i32x8;

// 2-D TDM load: tile_d0 contiguous 16-bit elements x tile_d1 rows, row stride
// `stride_elems` (in elements), into LDS at lds_byte_addr. Optional LDS row
// padding via pad_interval_code / pad_amount_code (ISA D# group1 encodings).
static __device__ __forceinline__ void tdm_load_2d_f16(
    unsigned lds_byte_addr, const void* gaddr, unsigned tile_d0,
    unsigned tile_d1, unsigned stride_elems, unsigned pad_interval_code,
    unsigned pad_amount_code, int pad_en) {
  unsigned long long ga = (unsigned long long)gaddr;
  u32x4 g0;
  g0[0] = 1u;                                     // count=1 (valid), user mode
  g0[1] = lds_byte_addr;                          // lds_addr
  g0[2] = (unsigned)(ga & 0xffffffffu);           // global_addr[31:0]
  g0[3] = (unsigned)((ga >> 32) & 0x01ffffffu)    // global_addr[56:32]
          | (2u << 30);                           // type=2 ("image")
  i32x8 g1;
  g1[0] = (int)((1u << 16)                        // data_size = 2 bytes
                | (pad_en ? (1u << 20) : 0u)      // pad_enable
                | (pad_interval_code << 22)       // pad every 2^(c+1) DWORDs
                | (pad_amount_code << 25));       // pad (c+1) DWORDs
  g1[1] = (int)((tile_d0 & 0xffffu) << 16);       // tensor_dim0[15:0] = tile_d0
  g1[2] = (int)(((tile_d0 >> 16) & 0xffffu) |     // tensor_dim0[31:16]
                ((tile_d1 & 0xffffu) << 16));     // tensor_dim1[15:0] = tile_d1
  g1[3] = (int)(((tile_d1 >> 16) & 0xffffu) |     // tensor_dim1[31:16]
                ((tile_d0 & 0xffffu) << 16));     // tile_dim0
  g1[4] = (int)(tile_d1 & 0xffffu);               // tile_dim1 ; tile_dim2 = 0
  g1[5] = (int)stride_elems;                      // tensor_dim0_stride[31:0]
  g1[6] = 0;                                      // stride hi / dim1_stride lo
  g1[7] = 0;
  i32x4 z4 = {0, 0, 0, 0};
#if defined(__clang_major__) && (__clang_major__ >= 23)
  i32x8 z8 = {0, 0, 0, 0, 0, 0, 0, 0};
  __builtin_amdgcn_tensor_load_to_lds(g0, g1, z4, z4, z8, 0);
#else
  __builtin_amdgcn_tensor_load_to_lds(g0, g1, z4, z4, 0);
#endif
}
#endif

// ======================================================================
// Generic WMMA GEMM:  C[M,N] = A[M,K] * W[K,N] + bias[N]
// A is f32 or f16 (template), W/bias f32, C is f16 or f32 (template).
// Workgroup tile 128x128, 8 waves, each wave 32x64 (2x4 WMMA tiles), BK=32.
// ======================================================================
template <bool A_HALF, bool OUT_HALF>
__global__ __launch_bounds__(256) void gemm_wmma_kernel(
    const void* __restrict__ Ain, const float* __restrict__ W,
    const float* __restrict__ bias, void* __restrict__ Cout,
    int M, int N, int K) {
  constexpr int BM = 128, BN = 128, BK = 32, LDA = 40, LDB = 40;
  __shared__ __align__(16) half_t As[BM * LDA];  // [m][k]
  __shared__ __align__(16) half_t Bs[BN * LDB];  // transposed: [n][k]

  const int tid  = threadIdx.x;
  const int lane = tid & 31;
  const int wid  = tid >> 5;
  const int wm   = wid & 3;   // wave row (32 rows each)
  const int wn   = wid >> 2;  // wave col (64 cols each)
  const int ln   = lane & 15;
  const int hf   = lane >> 4;

  const int m0 = blockIdx.y * BM;
  const int n0 = blockIdx.x * BN;

  v8f acc[2][4] = {};

  const int arow = tid >> 1;        // 0..127
  const int akp  = (tid & 1) * 16;  // 0 / 16
  const int bn4  = (tid & 31) * 4;  // 0..124
  const int bk0  = tid >> 5;        // 0..7 (rows bk0 + 8j)

  const float*  Af = (const float*)Ain;
  const half_t* Ah = (const half_t*)Ain;

  for (int k0 = 0; k0 < K; k0 += BK) {
    __syncthreads();
    // ---- stage A tile ----
    if (A_HALF) {
      const half_t* src = Ah + (size_t)(m0 + arow) * K + k0 + akp;
#pragma unroll
      for (int i = 0; i < 8; ++i)
        *(half2_t*)&As[arow * LDA + akp + 2 * i] = *(const half2_t*)(src + 2 * i);
    } else {
      const float* src = Af + (size_t)(m0 + arow) * K + k0 + akp;
#pragma unroll
      for (int i = 0; i < 4; ++i) {
        float4 f = *(const float4*)(src + 4 * i);
        half2_t p0 = {(half_t)f.x, (half_t)f.y};
        half2_t p1 = {(half_t)f.z, (half_t)f.w};
        *(half2_t*)&As[arow * LDA + akp + 4 * i]     = p0;
        *(half2_t*)&As[arow * LDA + akp + 4 * i + 2] = p1;
      }
    }
    // ---- stage B tile, transposed to [n][k] ----
#pragma unroll
    for (int j = 0; j < 4; ++j) {
      const int kk = bk0 + j * 8;
      float4 f = *(const float4*)(W + (size_t)(k0 + kk) * N + n0 + bn4);
      Bs[(bn4 + 0) * LDB + kk] = (half_t)f.x;
      Bs[(bn4 + 1) * LDB + kk] = (half_t)f.y;
      Bs[(bn4 + 2) * LDB + kk] = (half_t)f.z;
      Bs[(bn4 + 3) * LDB + kk] = (half_t)f.w;
    }
    // ---- prefetch next K tile sources (global_prefetch_b8) ----
    if (k0 + BK < K) {
      if (A_HALF)
        __builtin_prefetch(Ah + (size_t)(m0 + arow) * K + k0 + BK + akp, 0, 1);
      else
        __builtin_prefetch(Af + (size_t)(m0 + arow) * K + k0 + BK + akp, 0, 1);
      __builtin_prefetch(W + (size_t)(k0 + BK + bk0) * N + n0 + bn4, 0, 1);
      __builtin_prefetch(W + (size_t)(k0 + BK + bk0 + 16) * N + n0 + bn4, 0, 1);
    }
    __syncthreads();

    // ---- fragments + 8 WMMAs ----
    v16h afr[2], bfr[4];
#pragma unroll
    for (int sm = 0; sm < 2; ++sm) {
      const int row = wm * 32 + sm * 16 + ln;
#pragma unroll
      for (int v = 0; v < 8; ++v) {
        half2_t p = *(const half2_t*)&As[row * LDA + a_frag_kbase(v, hf)];
        afr[sm][2 * v] = p[0];
        afr[sm][2 * v + 1] = p[1];
      }
    }
#pragma unroll
    for (int sn = 0; sn < 4; ++sn) {
      const int col = wn * 64 + sn * 16 + ln;
#pragma unroll
      for (int i = 0; i < 8; ++i) {
        half2_t p = *(const half2_t*)&Bs[col * LDB + a_frag_kbase(i, hf)];
        bfr[sn][2 * i] = p[0];
        bfr[sn][2 * i + 1] = p[1];
      }
    }
#pragma unroll
    for (int sm = 0; sm < 2; ++sm)
#pragma unroll
      for (int sn = 0; sn < 4; ++sn)
        acc[sm][sn] = WMMA_F16(afr[sm], bfr[sn], acc[sm][sn]);
  }

  // ---- epilogue: bias + store (C layout: lane ln = col, rows r+8*hf) ----
#pragma unroll
  for (int sm = 0; sm < 2; ++sm) {
#pragma unroll
    for (int sn = 0; sn < 4; ++sn) {
      const int col = n0 + wn * 64 + sn * 16 + ln;
      const float bv = bias[col];
#pragma unroll
      for (int r = 0; r < 8; ++r) {
        const int row = m0 + wm * 32 + sm * 16 + r + 8 * hf;
        const float val = acc[sm][sn][r] + bv;
        if (OUT_HALF)
          ((half_t*)Cout)[(size_t)row * N + col] = (half_t)val;
        else
          ((float*)Cout)[(size_t)row * N + col] = val;
      }
    }
  }
}

// ======================================================================
// Flash attention: one workgroup per (b, h, 64-query block); 4 waves,
// 16 query rows per wave. qkv is f16 [B*S, 3072] (Q|K|V, col = h*64+dh).
// K tile is DMA'd into LDS by the Tensor Data Mover (with LDS row padding
// done by the TDM pad feature so LDK=72); V tile staged transposed by VALU.
// Causal masking runs only on the 1-2 diagonal key blocks (uniform branch).
// ======================================================================
__global__ __launch_bounds__(128) void attn_kernel(
    const half_t* __restrict__ qkv, half_t* __restrict__ attn,
    const int* __restrict__ causal_ptr) {
  constexpr int S = 2048, DQKV = 3072, Dh = 64;
  constexpr int LDK = 72, LDV = 34;
  __shared__ __align__(16) half_t Ks[32 * LDK];  // [key][dh], 8-half row pad
  __shared__ __align__(16) half_t Vs[64 * LDV];  // transposed: [dh][key]
  __shared__ __align__(16) half_t Pw[4 * 16 * 32];

  const int tid  = threadIdx.x;
  const int lane = tid & 31;
  const int wid  = tid >> 5;
  const int ln   = lane & 15;
  const int hf   = lane >> 4;

  const int b = blockIdx.y >> 4;
  const int h = blockIdx.y & 15;
  const int qblk  = blockIdx.x;
  const int qbase = qblk * 64 + wid * 16;
  const size_t rowbase = (size_t)b * S;
  const int causal = *causal_ptr;

  // ---- Q fragments (held in registers the whole pass) ----
  v16h aq[2];
  {
    const half_t* qrow = qkv + (rowbase + qbase + ln) * DQKV + h * Dh;
#pragma unroll
    for (int kc = 0; kc < 2; ++kc)
#pragma unroll
      for (int v = 0; v < 8; ++v) {
        half2_t p = *(const half2_t*)(qrow + kc * 32 + a_frag_kbase(v, hf));
        aq[kc][2 * v] = p[0];
        aq[kc][2 * v + 1] = p[1];
      }
  }

  float m_i[8], l_i[8];
#pragma unroll
  for (int r = 0; r < 8; ++r) { m_i[r] = -__builtin_inff(); l_i[r] = 0.0f; }
  v8f acc[4] = {};

  const float sc = 0.125f;  // 1/sqrt(64)
  const int skey = tid >> 2;        // 0..31
  const int sdh  = (tid & 3) * 16;  // 0,16,32,48

#ifdef HAVE_TDM
  const unsigned ks_lds = (unsigned)(unsigned long long)(void*)&Ks[0];
#endif

  const int nkb = causal ? (qblk * 2 + 2) : (S / 32);
  for (int kb = 0; kb < nkb; ++kb) {
    __syncthreads();
#ifdef HAVE_TDM
    // ---- TDM: DMA K tile [32 keys][64 dh] -> Ks (pad 4 DWORDs / 32 DWORDs
    //      => LDS row stride 72 halves). Issued by wave 0 only.
    if (wid == 0) {
      const half_t* ksrc = qkv + (rowbase + kb * 32) * DQKV + 1024 + h * Dh;
      tdm_load_2d_f16(ks_lds, ksrc, /*d0=*/64, /*d1=*/32, /*stride=*/DQKV,
                      /*pad_interval: 2^(4+1)=32 DW*/ 4,
                      /*pad_amount: 3+1=4 DW*/ 3, /*pad_en=*/1);
    }
#else
    // ---- fallback: stage K tile with vector loads ----
    {
      const half_t* src =
          qkv + (rowbase + kb * 32 + skey) * DQKV + 1024 + h * Dh + sdh;
      *(uint4*)&Ks[skey * LDK + sdh]     = *(const uint4*)(src);
      *(uint4*)&Ks[skey * LDK + sdh + 8] = *(const uint4*)(src + 8);
    }
#endif
    // ---- stage V tile transposed [64 dh][32 keys] ----
    {
      const half_t* src =
          qkv + (rowbase + kb * 32 + skey) * DQKV + 2048 + h * Dh + sdh;
#pragma unroll
      for (int j = 0; j < 16; ++j) Vs[(sdh + j) * LDV + skey] = src[j];
    }
    // prefetch next iteration's K/V rows
    if (kb + 1 < nkb) {
      __builtin_prefetch(
          qkv + (rowbase + (kb + 1) * 32 + skey) * DQKV + 1024 + h * Dh + sdh, 0, 1);
      __builtin_prefetch(
          qkv + (rowbase + (kb + 1) * 32 + skey) * DQKV + 2048 + h * Dh + sdh, 0, 1);
    }
#if defined(HAVE_TDM) && __has_builtin(__builtin_amdgcn_s_wait_tensorcnt)
    if (wid == 0) __builtin_amdgcn_s_wait_tensorcnt(0);
#endif
    __syncthreads();

    // ---- scores: S16x32 = Q16x64 * K^T, via 4 WMMAs ----
    v8f scv[2];
#pragma unroll
    for (int sn = 0; sn < 2; ++sn) {
      v8f c = {};
      const int key = sn * 16 + ln;
#pragma unroll
      for (int kc = 0; kc < 2; ++kc) {
        v16h bk;
#pragma unroll
        for (int i = 0; i < 8; ++i) {
          half2_t p = *(const half2_t*)&Ks[key * LDK + kc * 32 + a_frag_kbase(i, hf)];
          bk[2 * i] = p[0];
          bk[2 * i + 1] = p[1];
        }
        c = WMMA_F16(aq[kc], bk, c);
      }
      scv[sn] = c;
    }

    // ---- scale (+ mask only on diagonal-straddling blocks; wave-uniform) ----
    float s0[8], s1[8], pscale[8];
    const bool need_mask = causal && (kb * 32 + 31 > qbase);
    if (need_mask) {
#pragma unroll
      for (int r = 0; r < 8; ++r) {
        const int q   = qbase + r + 8 * hf;
        const int k0i = kb * 32 + ln;
        const int k1i = kb * 32 + 16 + ln;
        s0[r] = (k0i <= q) ? scv[0][r] * sc : -__builtin_inff();
        s1[r] = (k1i <= q) ? scv[1][r] * sc : -__builtin_inff();
      }
    } else {
#pragma unroll
      for (int r = 0; r < 8; ++r) {
        s0[r] = scv[0][r] * sc;
        s1[r] = scv[1][r] * sc;
      }
    }
    // ---- online softmax ----
#pragma unroll
    for (int r = 0; r < 8; ++r) {
      float t = fmaxf(s0[r], s1[r]);
#pragma unroll
      for (int msk = 1; msk < 16; msk <<= 1)
        t = fmaxf(t, __shfl_xor(t, msk, 32));  // reduce within 16-lane row group
      const float mn = fmaxf(m_i[r], t);
      pscale[r] = __expf(m_i[r] - mn);
      s0[r] = __expf(s0[r] - mn);
      s1[r] = __expf(s1[r] - mn);
      float rsum = s0[r] + s1[r];
#pragma unroll
      for (int msk = 1; msk < 16; msk <<= 1)
        rsum += __shfl_xor(rsum, msk, 32);
      l_i[r] = l_i[r] * pscale[r] + rsum;
      m_i[r] = mn;
    }
#pragma unroll
    for (int t = 0; t < 4; ++t)
#pragma unroll
      for (int r = 0; r < 8; ++r) acc[t][r] *= pscale[r];

    // ---- P: C-layout -> A-layout via per-wave LDS tile ----
    half_t* P = &Pw[wid * 16 * 32];
#pragma unroll
    for (int r = 0; r < 8; ++r) {
      P[(r + 8 * hf) * 32 + ln]      = (half_t)s0[r];
      P[(r + 8 * hf) * 32 + 16 + ln] = (half_t)s1[r];
    }
    __syncthreads();

    v16h pf;
#pragma unroll
    for (int v = 0; v < 8; ++v) {
      half2_t p = *(const half2_t*)&P[ln * 32 + a_frag_kbase(v, hf)];
      pf[2 * v] = p[0];
      pf[2 * v + 1] = p[1];
    }
    // ---- O += P * V : 4 WMMAs over the 64 dh columns ----
#pragma unroll
    for (int nb = 0; nb < 4; ++nb) {
      v16h bv;
      const int dh = nb * 16 + ln;
#pragma unroll
      for (int i = 0; i < 8; ++i) {
        half2_t p = *(const half2_t*)&Vs[dh * LDV + a_frag_kbase(i, hf)];
        bv[2 * i] = p[0];
        bv[2 * i + 1] = p[1];
      }
      acc[nb] = WMMA_F16(pf, bv, acc[nb]);
    }
  }

  // ---- normalize + store attention output (f16 workspace) ----
#pragma unroll
  for (int nb = 0; nb < 4; ++nb) {
#pragma unroll
    for (int r = 0; r < 8; ++r) {
      const int s = qbase + r + 8 * hf;
      const float v = acc[nb][r] / l_i[r];
      attn[(rowbase + s) * 1024 + h * Dh + nb * 16 + ln] = (half_t)v;
    }
  }
}

// ======================================================================
extern "C" void kernel_launch(void* const* d_in, const int* in_sizes, int n_in,
                              void* d_out, int out_size, void* d_ws,
                              size_t ws_size, hipStream_t stream) {
  const float* x      = (const float*)d_in[0];
  const float* w_in   = (const float*)d_in[1];
  const float* b_in   = (const float*)d_in[2];
  const float* w_out  = (const float*)d_in[3];
  const float* b_out  = (const float*)d_in[4];
  const int*   causal = (const int*)d_in[5];

  constexpr int BS = 4 * 2048;  // 8192 rows
  half_t* qkv  = (half_t*)d_ws;                                   // 8192x3072 f16
  half_t* attn = (half_t*)((char*)d_ws + (size_t)BS * 3072 * 2);  // 8192x1024 f16

  // 1) QKV projection: f32 in -> f16 out
  gemm_wmma_kernel<false, true>
      <<<dim3(3072 / 128, BS / 128), 256, 0, stream>>>(x, w_in, b_in, qkv,
                                                       BS, 3072, 1024);
  // 2) causal flash attention (f16 in/out, f32 accumulate)
  attn_kernel<<<dim3(2048 / 64, 4 * 16), 128, 0, stream>>>(qkv, attn, causal);

  // 3) output projection: f16 in -> f32 out
  gemm_wmma_kernel<true, false>
      <<<dim3(1024 / 128, BS / 128), 256, 0, stream>>>(attn, w_out, b_out,
                                                       (float*)d_out,
                                                       BS, 1024, 1024);
}
